// MoLELayer_21457656611048
// MI455X (gfx1250) — compile-verified, matrix operand order checked by reading, not run on hardware
//
#include <hip/hip_runtime.h>
#include <math.h>

// ---------------------------------------------------------------------------
// Types for CDNA5 WMMA (gfx1250, wave32)
// ---------------------------------------------------------------------------
typedef __bf16 bf16_t;
typedef bf16_t v16bf __attribute__((ext_vector_type(16)));
typedef bf16_t v8bf  __attribute__((ext_vector_type(8)));
typedef float  v8f   __attribute__((ext_vector_type(8)));

union BF16x16 { v16bf v; v8bf h[2]; };

#define BM 128
#define BN 256
#define BK 32
#define LDS_STRIDE 40   // 32 bf16 + 8 bf16 pad (16B); keeps every k-offset 16B-aligned

__device__ __forceinline__ unsigned short f2bf(float f) {
  union { float f; unsigned u; } c; c.f = f;
  unsigned u = c.u;
  unsigned r = u + 0x7FFFu + ((u >> 16) & 1u);   // round-to-nearest-even
  return (unsigned short)(r >> 16);
}

__device__ __forceinline__ float gelu_tanh(float v) {
  return 0.5f * v * (1.0f + tanhf(0.7978845608028654f * (v + 0.044715f * v * v * v)));
}

// ---------------------------------------------------------------------------
// Double-buffered WMMA mainloop: C(128x256) += A[M,K](bf16) * Bt[N,K]^T
// Bt stored N-major ([N][K]) so A and B tiles load identically (16B coalesced).
// 256 threads = 8 waves, 2x4 wave grid, each wave owns 64x64 -> 16 WMMA/K-step.
// One barrier per K-step (buf[k&1] read, buf[(k+1)&1] written).
// ---------------------------------------------------------------------------
__device__ __forceinline__ void mma_mainloop(
    const bf16_t* __restrict__ A, const bf16_t* __restrict__ Bt,
    int K, int blockM, int blockN,
    bf16_t* sA, bf16_t* sB, v8f acc[4][4])
{
  const int tid    = threadIdx.x;
  const int lane   = tid & 31;
  const int wave   = tid >> 5;
  const int wmBase = (wave & 1) * 64;
  const int wnBase = (wave >> 1) * 64;
  const int lrow   = lane & 15;
  const int koff   = (lane >> 4) * 8;   // K-half select per CDNA5 16-bit operand layout

  const int lr = tid >> 2;              // 0..63 : tile row
  const int lc = (tid & 3) * 8;         // 0,8,16,24 : tile col (bf16 elems)

  const int ABUF = BM * LDS_STRIDE;
  const int BBUF = BN * LDS_STRIDE;

  const bf16_t* gA = A  + (size_t)(blockM + lr) * K + lc;
  const bf16_t* gB = Bt + (size_t)(blockN + lr) * K + lc;

  v8bf ra0, ra1, rb0, rb1, rb2, rb3;

  // ---- prologue: tile 0 -> buf0 ----
  ra0 = *(const v8bf*)(gA);
  ra1 = *(const v8bf*)(gA + (size_t)64  * K);
  rb0 = *(const v8bf*)(gB);
  rb1 = *(const v8bf*)(gB + (size_t)64  * K);
  rb2 = *(const v8bf*)(gB + (size_t)128 * K);
  rb3 = *(const v8bf*)(gB + (size_t)192 * K);
  *(v8bf*)(sA + (size_t)lr         * LDS_STRIDE + lc) = ra0;
  *(v8bf*)(sA + (size_t)(lr + 64)  * LDS_STRIDE + lc) = ra1;
  *(v8bf*)(sB + (size_t)lr         * LDS_STRIDE + lc) = rb0;
  *(v8bf*)(sB + (size_t)(lr + 64)  * LDS_STRIDE + lc) = rb1;
  *(v8bf*)(sB + (size_t)(lr + 128) * LDS_STRIDE + lc) = rb2;
  *(v8bf*)(sB + (size_t)(lr + 192) * LDS_STRIDE + lc) = rb3;
  __syncthreads();

  const int nk = K / BK;
  for (int k = 0; k < nk; ++k) {
    const bf16_t* cA = sA + (size_t)(k & 1) * ABUF;
    const bf16_t* cB = sB + (size_t)(k & 1) * BBUF;

    // ---- fetch next tile into registers (overlaps WMMA chain) ----
    if (k + 1 < nk) {
      const int ko = (k + 1) * BK;
      ra0 = *(const v8bf*)(gA + ko);
      ra1 = *(const v8bf*)(gA + (size_t)64  * K + ko);
      rb0 = *(const v8bf*)(gB + ko);
      rb1 = *(const v8bf*)(gB + (size_t)64  * K + ko);
      rb2 = *(const v8bf*)(gB + (size_t)128 * K + ko);
      rb3 = *(const v8bf*)(gB + (size_t)192 * K + ko);
      __builtin_prefetch(gA + ko + BK, 0, 1);   // k+2 tile -> global_prefetch_b8
      __builtin_prefetch(gB + ko + BK, 0, 1);
    }

    // ---- fragments from current buffer ----
    BF16x16 fa[4];
    #pragma unroll
    for (int mt = 0; mt < 4; ++mt) {
      const bf16_t* p = cA + (size_t)(wmBase + mt * 16 + lrow) * LDS_STRIDE + koff;
      fa[mt].h[0] = *(const v8bf*)p;
      fa[mt].h[1] = *(const v8bf*)(p + 16);
    }

    // ---- 16 WMMAs per K-step per wave ----
    #pragma unroll
    for (int nt = 0; nt < 4; ++nt) {
      BF16x16 fb;
      const bf16_t* p = cB + (size_t)(wnBase + nt * 16 + lrow) * LDS_STRIDE + koff;
      fb.h[0] = *(const v8bf*)p;
      fb.h[1] = *(const v8bf*)(p + 16);
      #pragma unroll
      for (int mt = 0; mt < 4; ++mt)
        acc[mt][nt] = __builtin_amdgcn_wmma_f32_16x16x32_bf16(
            false, fa[mt].v, false, fb.v, (short)0, acc[mt][nt], false, false);
    }

    // ---- stage next tile into the other buffer ----
    if (k + 1 < nk) {
      bf16_t* nA = sA + (size_t)((k + 1) & 1) * ABUF;
      bf16_t* nB = sB + (size_t)((k + 1) & 1) * BBUF;
      *(v8bf*)(nA + (size_t)lr         * LDS_STRIDE + lc) = ra0;
      *(v8bf*)(nA + (size_t)(lr + 64)  * LDS_STRIDE + lc) = ra1;
      *(v8bf*)(nB + (size_t)lr         * LDS_STRIDE + lc) = rb0;
      *(v8bf*)(nB + (size_t)(lr + 64)  * LDS_STRIDE + lc) = rb1;
      *(v8bf*)(nB + (size_t)(lr + 128) * LDS_STRIDE + lc) = rb2;
      *(v8bf*)(nB + (size_t)(lr + 192) * LDS_STRIDE + lc) = rb3;
    }
    __syncthreads();   // protects next reads of buf[(k+1)&1] AND reuse of buf[k&1]
  }
}

// ---------------------------------------------------------------------------
// GEMM1: H = gelu(A @ W1^T + b1), H stored bf16 row-major [M,N]
// ---------------------------------------------------------------------------
__global__ __launch_bounds__(256) void gemm1_gelu_bf16(
    const unsigned short* __restrict__ Au, const unsigned short* __restrict__ Btu,
    const float* __restrict__ bias, unsigned short* __restrict__ Hu,
    int M, int N, int K)
{
  __shared__ __align__(16) bf16_t sA[2 * BM * LDS_STRIDE];
  __shared__ __align__(16) bf16_t sB[2 * BN * LDS_STRIDE];

  const bf16_t* A  = (const bf16_t*)Au;
  const bf16_t* Bt = (const bf16_t*)Btu;
  const int blockN = blockIdx.x * BN;
  const int blockM = blockIdx.y * BM;

  v8f acc[4][4];
  #pragma unroll
  for (int mt = 0; mt < 4; ++mt)
    #pragma unroll
    for (int nt = 0; nt < 4; ++nt)
      #pragma unroll
      for (int i = 0; i < 8; ++i) acc[mt][nt][i] = 0.0f;

  mma_mainloop(A, Bt, K, blockM, blockN, sA, sB, acc);

  const int tid    = threadIdx.x;
  const int lane   = tid & 31;
  const int wave   = tid >> 5;
  const int wmBase = (wave & 1) * 64;
  const int wnBase = (wave >> 1) * 64;
  const int lrow   = lane & 15;
  const int mhalf  = (lane >> 4) * 8;

  #pragma unroll
  for (int mt = 0; mt < 4; ++mt) {
    #pragma unroll
    for (int nt = 0; nt < 4; ++nt) {
      const int col = blockN + wnBase + nt * 16 + lrow;
      const float bv = bias[col];
      #pragma unroll
      for (int i = 0; i < 8; ++i) {
        const int m = blockM + wmBase + mt * 16 + mhalf + i;
        const float v = acc[mt][nt][i] + bv;
        Hu[(size_t)m * N + col] = f2bf(gelu_tanh(v));
      }
    }
  }
}

// ---------------------------------------------------------------------------
// GEMM2: Out (+)= r[:,e] * (A @ W2^T + b2).  Rptr==nullptr -> write mode, w=1.
// ---------------------------------------------------------------------------
__global__ __launch_bounds__(256) void gemm2_routed(
    const unsigned short* __restrict__ Au, const unsigned short* __restrict__ Btu,
    const float* __restrict__ bias, const float* __restrict__ Rptr,
    int expert, int nExperts, float* __restrict__ Out,
    int M, int N, int K)
{
  __shared__ __align__(16) bf16_t sA[2 * BM * LDS_STRIDE];
  __shared__ __align__(16) bf16_t sB[2 * BN * LDS_STRIDE];

  const bf16_t* A  = (const bf16_t*)Au;
  const bf16_t* Bt = (const bf16_t*)Btu;
  const int blockN = blockIdx.x * BN;
  const int blockM = blockIdx.y * BM;

  v8f acc[4][4];
  #pragma unroll
  for (int mt = 0; mt < 4; ++mt)
    #pragma unroll
    for (int nt = 0; nt < 4; ++nt)
      #pragma unroll
      for (int i = 0; i < 8; ++i) acc[mt][nt][i] = 0.0f;

  mma_mainloop(A, Bt, K, blockM, blockN, sA, sB, acc);

  const int tid    = threadIdx.x;
  const int lane   = tid & 31;
  const int wave   = tid >> 5;
  const int wmBase = (wave & 1) * 64;
  const int wnBase = (wave >> 1) * 64;
  const int lrow   = lane & 15;
  const int mhalf  = (lane >> 4) * 8;

  #pragma unroll
  for (int mt = 0; mt < 4; ++mt) {
    #pragma unroll
    for (int nt = 0; nt < 4; ++nt) {
      const int col = blockN + wnBase + nt * 16 + lrow;
      const float bv = bias[col];
      #pragma unroll
      for (int i = 0; i < 8; ++i) {
        const int m = blockM + wmBase + mt * 16 + mhalf + i;
        const size_t o = (size_t)m * N + col;
        const float v = acc[mt][nt][i] + bv;
        if (Rptr) {
          const float rv = Rptr[(size_t)m * nExperts + expert];
          Out[o] += rv * v;
        } else {
          Out[o] = v;            // shared expert writes first (initializes d_out)
        }
      }
    }
  }
}

// ---------------------------------------------------------------------------
// Router: r = softmax(x @ rw + rb) over E=8.  One thread per (token, expert).
// ---------------------------------------------------------------------------
__global__ __launch_bounds__(256) void router_softmax(
    const float* __restrict__ x, const float* __restrict__ rw,
    const float* __restrict__ rb, float* __restrict__ r,
    int Ntok, int Hd, int E)
{
  const int gid = blockIdx.x * blockDim.x + threadIdx.x;
  const int tok = gid / E;
  const int e   = gid % E;
  if (tok >= Ntok) return;

  float a = rb[e];
  const float* xr = x + (size_t)tok * Hd;
  for (int k = 0; k < Hd; ++k) a += xr[k] * rw[(size_t)k * E + e];

  float mx = a;
  for (int off = 1; off < E; off <<= 1) mx = fmaxf(mx, __shfl_xor(mx, off, E));
  const float ex = __expf(a - mx);
  float s = ex;
  for (int off = 1; off < E; off <<= 1) s += __shfl_xor(s, off, E);
  r[(size_t)tok * E + e] = ex / s;
}

// ---------------------------------------------------------------------------
// f32 -> bf16 flat conversion
// ---------------------------------------------------------------------------
__global__ __launch_bounds__(256) void cvt_f32_bf16(
    const float* __restrict__ in, unsigned short* __restrict__ out, size_t n)
{
  const size_t i = blockIdx.x * (size_t)blockDim.x + threadIdx.x;
  if (i < n) out[i] = f2bf(in[i]);
}

// ---------------------------------------------------------------------------
// f32 [Z][R][C] -> bf16 [Z][C][R] tiled transpose (32x32 tiles, LDS)
// ---------------------------------------------------------------------------
__global__ __launch_bounds__(256) void transpose_f32_bf16(
    const float* __restrict__ in, unsigned short* __restrict__ out, int R, int C)
{
  __shared__ float tile[32][33];
  const int tx = threadIdx.x, ty = threadIdx.y;
  const int r0 = blockIdx.y * 32, c0 = blockIdx.x * 32;
  const float*    inp  = in  + (size_t)blockIdx.z * R * C;
  unsigned short* outp = out + (size_t)blockIdx.z * R * C;

  #pragma unroll
  for (int j = 0; j < 4; ++j)
    tile[ty + j * 8][tx] = inp[(size_t)(r0 + ty + j * 8) * C + c0 + tx];
  __syncthreads();
  #pragma unroll
  for (int j = 0; j < 4; ++j)
    outp[(size_t)(c0 + ty + j * 8) * R + r0 + tx] = f2bf(tile[tx][ty + j * 8]);
}

// ---------------------------------------------------------------------------
// Host orchestration
// ---------------------------------------------------------------------------
extern "C" void kernel_launch(void* const* d_in, const int* in_sizes, int n_in,
                              void* d_out, int out_size, void* d_ws, size_t ws_size,
                              hipStream_t stream) {
  (void)in_sizes; (void)n_in; (void)out_size; (void)ws_size;

  const float* x   = (const float*)d_in[0];
  const float* emb = (const float*)d_in[1];
  const float* rw  = (const float*)d_in[2];
  const float* rb  = (const float*)d_in[3];
  const float* sw1 = (const float*)d_in[4];
  const float* sb1 = (const float*)d_in[5];
  const float* sw2 = (const float*)d_in[6];
  const float* sb2 = (const float*)d_in[7];
  const float* ew1 = (const float*)d_in[8];
  const float* eb1 = (const float*)d_in[9];
  const float* ew2 = (const float*)d_in[10];
  const float* eb2 = (const float*)d_in[11];
  float* out = (float*)d_out;

  const int Bsz = 4, T = 1024, Hd = 768, E = 8, F = 3072;
  const int M = Bsz * T;                       // 4096 tokens

  // --- carve workspace (256B-aligned chunks) ---
  char* p = (char*)d_ws;
  auto carve = [&](size_t bytes) -> void* {
    void* q = (void*)p; p += (bytes + 255) & ~(size_t)255; return q;
  };
  unsigned short* x_bf   = (unsigned short*)carve((size_t)M * Hd * 2);
  unsigned short* emb_bf = (unsigned short*)carve((size_t)M * Hd * 2);
  unsigned short* sw1t   = (unsigned short*)carve((size_t)Hd * F * 2);     // [F][Hd]
  unsigned short* sw2t   = (unsigned short*)carve((size_t)Hd * F * 2);     // [Hd][F]
  unsigned short* ew1t   = (unsigned short*)carve((size_t)E * Hd * F * 2); // [E][F][Hd]
  unsigned short* ew2t   = (unsigned short*)carve((size_t)E * Hd * F * 2); // [E][Hd][F]
  unsigned short* hbuf   = (unsigned short*)carve((size_t)M * F * 2);      // [M][F] bf16
  float*          rbuf   = (float*)carve((size_t)M * E * 4);               // [M][E]

  // --- conversions (activations) ---
  {
    const size_t n = (size_t)M * Hd;
    const int blk = 256, grd = (int)((n + blk - 1) / blk);
    cvt_f32_bf16<<<grd, blk, 0, stream>>>(x, x_bf, n);
    cvt_f32_bf16<<<grd, blk, 0, stream>>>(emb, emb_bf, n);
  }
  // --- weight transposes to [N][K] bf16 ---
  transpose_f32_bf16<<<dim3(F / 32, Hd / 32, 1), dim3(32, 8), 0, stream>>>(sw1, sw1t, Hd, F);
  transpose_f32_bf16<<<dim3(Hd / 32, F / 32, 1), dim3(32, 8), 0, stream>>>(sw2, sw2t, F, Hd);
  transpose_f32_bf16<<<dim3(F / 32, Hd / 32, E), dim3(32, 8), 0, stream>>>(ew1, ew1t, Hd, F);
  transpose_f32_bf16<<<dim3(Hd / 32, F / 32, E), dim3(32, 8), 0, stream>>>(ew2, ew2t, F, Hd);

  // --- router softmax ---
  router_softmax<<<(M * E) / 256, 256, 0, stream>>>(x, rw, rb, rbuf, M, Hd, E);

  const dim3 g1(F / BN, M / BM);   // (12, 32)
  const dim3 g2(Hd / BN, M / BM);  // (3, 32)

  // --- shared expert: writes d_out (initializes it) ---
  gemm1_gelu_bf16<<<g1, 256, 0, stream>>>(x_bf, sw1t, sb1, hbuf, M, F, Hd);
  gemm2_routed  <<<g2, 256, 0, stream>>>(hbuf, sw2t, sb2, nullptr, 0, E, out, M, Hd, F);

  // --- routed experts: accumulate r[:,e] * MLP_e(emb) into d_out ---
  for (int e = 0; e < E; ++e) {
    const unsigned short* w1t = ew1t + (size_t)e * Hd * F;
    const unsigned short* w2t = ew2t + (size_t)e * Hd * F;
    gemm1_gelu_bf16<<<g1, 256, 0, stream>>>(emb_bf, w1t, eb1 + (size_t)e * F, hbuf, M, F, Hd);
    gemm2_routed  <<<g2, 256, 0, stream>>>(hbuf, w2t, eb2 + (size_t)e * Hd, rbuf, e, E, out, M, Hd, F);
  }
}
